// CrossAttention_82987358093517
// MI455X (gfx1250) — compile-verified, matrix operand order checked by reading, not run on hardware
//
#include <hip/hip_runtime.h>
#include <hip/hip_bf16.h>

// ---------------------------------------------------------------------------
// CDNA5 (gfx1250) fused cross-attention, f16 WMMA path, f32 accumulate.
// B=4, N=1024, C=1024, H=16, HD=64
// GEMMs: 32x64 register tile per wave -> 8 v_wmma per 32-K chunk at
// 1.5 global_load_b128 per wmma (was 4.0).
// ---------------------------------------------------------------------------

typedef __attribute__((ext_vector_type(16))) _Float16 half16;
typedef __attribute__((ext_vector_type(8)))  _Float16 half8;
typedef __attribute__((ext_vector_type(4)))  _Float16 half4;
typedef __attribute__((ext_vector_type(8)))  float    float8;
typedef __attribute__((ext_vector_type(4)))  float    float4_t;

#define WMMA_F16(a, b, c) \
    __builtin_amdgcn_wmma_f32_16x16x32_f16(false, (a), false, (b), (short)0, (c), false, false)

// ---------------------------------------------------------------------------
// Fragment loader: 16-bit A/B fragment for v_wmma_f32_16x16x32_f16.
// Per ISA 7.12.2: lane l<16 holds "row" l, K = {c..c+7, c+16..c+23} with
// c = 0 for lanes 0-15 and c = 8 for lanes 16-31.  `base` points at
// (row0, k0) of a row-major f16 matrix with leading dim `ld` (elements).
// Rows of A and rows of B^T (i.e. columns of B) use the same loader.
// ---------------------------------------------------------------------------
__device__ __forceinline__ half16 frag_load(const _Float16* base, int ld) {
    const int lane = threadIdx.x & 31;
    const _Float16* p = base + (size_t)(lane & 15) * ld + ((lane >> 4) << 3);
    half8 lo = *(const half8*)p;
    half8 hi = *(const half8*)(p + 16);
    half16 r;
#pragma unroll
    for (int i = 0; i < 8; ++i) { r[i] = lo[i]; r[i + 8] = hi[i]; }
    return r;
}

// ---------------------------------------------------------------------------
// f32 -> f16 convert (vectorized by 4)
// ---------------------------------------------------------------------------
__global__ void ca_cvt_f16(const float* __restrict__ in, _Float16* __restrict__ out, int n4) {
    int i = blockIdx.x * blockDim.x + threadIdx.x;
    if (i >= n4) return;
    float4_t v = *(const float4_t*)(in + (size_t)i * 4);
    half4 o;
#pragma unroll
    for (int j = 0; j < 4; ++j) o[j] = (_Float16)v[j];
    *(half4*)(out + (size_t)i * 4) = o;
}

// ---------------------------------------------------------------------------
// Weight prep: W f32 [K][N] row-major  ->  Wt f16 [N][K] row-major.
// 32x32 LDS tile, block 32x8.
// ---------------------------------------------------------------------------
__global__ void ca_transpose_f16(const float* __restrict__ W, _Float16* __restrict__ Wt,
                                 int K, int N) {
    __shared__ float tile[32][33];
    const int bx = blockIdx.x;   // along N
    const int by = blockIdx.y;   // along K
    const int tx = threadIdx.x;  // 0..31
    const int ty = threadIdx.y;  // 0..7
#pragma unroll
    for (int i = 0; i < 4; ++i)
        tile[ty + i * 8][tx] = W[(size_t)(by * 32 + ty + i * 8) * N + bx * 32 + tx];
    __syncthreads();
#pragma unroll
    for (int i = 0; i < 4; ++i)
        Wt[(size_t)(bx * 32 + ty + i * 8) * K + by * 32 + tx] = (_Float16)tile[tx][ty + i * 8];
}

// ---------------------------------------------------------------------------
// WMMA GEMM mainloop, 32x64 per wave:
//   C[32][64] += A[32][K] * Bt[64][K]^T
// acc laid out as acc[i][j]: i in {0,1} = 16-row tile, j in {0..3} = 16-col
// tile.  Per 32-K chunk: 2 A-frag + 4 B-frag loads (12 x b128), 8 wmma.
// ---------------------------------------------------------------------------
__device__ __forceinline__ void gemm_mainloop_32x64(const _Float16* Arow, int lda,
                                                    const _Float16* Bcol, int ldb, int K,
                                                    float8 acc[2][4]) {
    for (int kc = 0; kc < K; kc += 32) {
        half16 a0 = frag_load(Arow + kc, lda);
        half16 a1 = frag_load(Arow + (size_t)16 * lda + kc, lda);
        half16 b0 = frag_load(Bcol + kc, ldb);
        half16 b1 = frag_load(Bcol + (size_t)16 * ldb + kc, ldb);
        half16 b2 = frag_load(Bcol + (size_t)32 * ldb + kc, ldb);
        half16 b3 = frag_load(Bcol + (size_t)48 * ldb + kc, ldb);
        acc[0][0] = WMMA_F16(a0, b0, acc[0][0]);
        acc[0][1] = WMMA_F16(a0, b1, acc[0][1]);
        acc[0][2] = WMMA_F16(a0, b2, acc[0][2]);
        acc[0][3] = WMMA_F16(a0, b3, acc[0][3]);
        acc[1][0] = WMMA_F16(a1, b0, acc[1][0]);
        acc[1][1] = WMMA_F16(a1, b1, acc[1][1]);
        acc[1][2] = WMMA_F16(a1, b2, acc[1][2]);
        acc[1][3] = WMMA_F16(a1, b3, acc[1][3]);
    }
}

// Block: 128 threads (4 waves); wave w owns rows [bx*128 + w*32, +32),
// block owns cols [by*64, +64).  Grid: (M/128, N/64).

// Epilogue variant 0: f16 row-major store
__global__ void __launch_bounds__(128)
ca_gemm_f16(const _Float16* __restrict__ A, const _Float16* __restrict__ Bt,
            _Float16* __restrict__ C, int lda, int ldb, int ldc, int K) {
    const int wave = threadIdx.x >> 5;
    const int lane = threadIdx.x & 31;
    const int row0 = blockIdx.x * 128 + wave * 32;
    const int col0 = blockIdx.y * 64;
    const int sub = lane & 15, hi8 = (lane >> 4) * 8;
    float8 acc[2][4] = {};
    gemm_mainloop_32x64(A + (size_t)row0 * lda, lda, Bt + (size_t)col0 * ldb, ldb, K, acc);
#pragma unroll
    for (int i = 0; i < 2; ++i)
#pragma unroll
        for (int j = 0; j < 4; ++j)
#pragma unroll
            for (int r = 0; r < 8; ++r)
                C[(size_t)(row0 + 16 * i + hi8 + r) * ldc + col0 + 16 * j + sub] =
                    (_Float16)acc[i][j][r];
}

// Epilogue variant 1: V-transposed store.  GEMM rows are (b*1024+key), cols
// j=h*64+d of the V half.  Store Vt[((b*16+h)*64+d)][key]: each lane owns one
// (h,d) column and 8 consecutive keys -> single v8h (16B) store per tile.
__global__ void __launch_bounds__(128)
ca_gemm_vstore(const _Float16* __restrict__ A, const _Float16* __restrict__ Bt,
               _Float16* __restrict__ Vt, int lda, int ldb, int K) {
    const int wave = threadIdx.x >> 5;
    const int lane = threadIdx.x & 31;
    const int row0 = blockIdx.x * 128 + wave * 32;
    const int col0 = blockIdx.y * 64;
    const int sub = lane & 15, hi8 = (lane >> 4) * 8;
    float8 acc[2][4] = {};
    gemm_mainloop_32x64(A + (size_t)row0 * lda, lda, Bt + (size_t)col0 * ldb, ldb, K, acc);
#pragma unroll
    for (int i = 0; i < 2; ++i) {
        const int row = row0 + 16 * i;
        const int b = row >> 10, key0 = row & 1023;
#pragma unroll
        for (int j = 0; j < 4; ++j) {
            const int col = col0 + 16 * j + sub;  // 0..1023 within V half
            const int h = col >> 6, d = col & 63;
            half8 o;
#pragma unroll
            for (int r = 0; r < 8; ++r) o[r] = (_Float16)acc[i][j][r];
            *(half8*)(Vt + (size_t)((b * 16 + h) * 64 + d) * 1024 + key0 + hi8) = o;
        }
    }
}

// Epilogue variant 2: f32 store + bias (final projection)
__global__ void __launch_bounds__(128)
ca_gemm_f32_bias(const _Float16* __restrict__ A, const _Float16* __restrict__ Bt,
                 float* __restrict__ C, const float* __restrict__ bias,
                 int lda, int ldb, int ldc, int K) {
    const int wave = threadIdx.x >> 5;
    const int lane = threadIdx.x & 31;
    const int row0 = blockIdx.x * 128 + wave * 32;
    const int col0 = blockIdx.y * 64;
    const int sub = lane & 15, hi8 = (lane >> 4) * 8;
    float8 acc[2][4] = {};
    gemm_mainloop_32x64(A + (size_t)row0 * lda, lda, Bt + (size_t)col0 * ldb, ldb, K, acc);
#pragma unroll
    for (int j = 0; j < 4; ++j) {
        const float bs = bias[col0 + 16 * j + sub];
#pragma unroll
        for (int i = 0; i < 2; ++i)
#pragma unroll
            for (int r = 0; r < 8; ++r)
                C[(size_t)(row0 + 16 * i + hi8 + r) * ldc + col0 + 16 * j + sub] =
                    acc[i][j][r] + bs;
    }
}

// ---------------------------------------------------------------------------
// Fused attention: per block one (b, h, 16-query tile).
// 4 waves / 128 threads.  LDS: S[16][1024] f32 (scores, 64KB) +
// P[16][1040] f16 (probabilities, padded rows: 2080B stride = 16B aligned,
// bank-shift 8 per row).  Total dynamic LDS = 98816 B -> 3 blocks / WGP.
// ---------------------------------------------------------------------------
#define PLD 1040
__device__ __forceinline__ float wred_max(float v) {
#pragma unroll
    for (int o = 16; o > 0; o >>= 1) v = fmaxf(v, __shfl_xor(v, o));
    return v;
}
__device__ __forceinline__ float wred_sum(float v) {
#pragma unroll
    for (int o = 16; o > 0; o >>= 1) v += __shfl_xor(v, o);
    return v;
}

__global__ void __launch_bounds__(128)
ca_attention(const _Float16* __restrict__ Qh, const _Float16* __restrict__ Kh,
             const _Float16* __restrict__ Vt, const float* __restrict__ attn_add,
             _Float16* __restrict__ AOut) {
    extern __shared__ char smem[];
    float*    S = (float*)smem;                       // [16][1024]
    _Float16* P = (_Float16*)(smem + 16 * 1024 * 4);  // [16][PLD]

    const int idx  = blockIdx.x;
    const int n0   = (idx & 63) * 16;
    const int h    = (idx >> 6) & 15;
    const int b    = idx >> 10;
    const int wave = threadIdx.x >> 5;
    const int lane = threadIdx.x & 31;
    const int sub  = lane & 15, hi8 = (lane >> 4) * 8;
    const float SCALE = 0.125f;  // 64^-0.5

    // Q fragments (16 queries x 64 dims), reused for all 64 key tiles.
    const _Float16* Qbase = Qh + (size_t)(b * 1024 + n0) * 1024 + h * 64;
    half16 qa0 = frag_load(Qbase + 0, 1024);
    half16 qa1 = frag_load(Qbase + 32, 1024);

    // ---- scores = Q K^T * scale -> LDS --------------------------------
    for (int t = wave; t < 64; t += 4) {
        const _Float16* Kbase = Kh + (size_t)(b * 1024 + t * 16) * 1024 + h * 64;
        half16 kb0 = frag_load(Kbase + 0, 1024);
        half16 kb1 = frag_load(Kbase + 32, 1024);
        float8 acc = {};
        acc = WMMA_F16(qa0, kb0, acc);
        acc = WMMA_F16(qa1, kb1, acc);
#pragma unroll
        for (int r = 0; r < 8; ++r)
            S[(hi8 + r) * 1024 + t * 16 + sub] = acc[r] * SCALE;
    }
    __syncthreads();

    // ---- softmax -> *attn_add -> softmax, per row (4 rows per wave) ----
    for (int rr = 0; rr < 4; ++rr) {
        const int r = wave * 4 + rr;
        float* Sr = S + r * 1024;
        const float* add = attn_add + ((size_t)b * 1024 + n0 + r) * 1024;

        float m1 = -1e30f;
        for (int c = lane; c < 1024; c += 32) m1 = fmaxf(m1, Sr[c]);
        m1 = wred_max(m1);
        float s1 = 0.f;
        for (int c = lane; c < 1024; c += 32) { float e = __expf(Sr[c] - m1); Sr[c] = e; s1 += e; }
        s1 = wred_sum(s1);
        const float inv1 = __frcp_rn(s1);

        float m2 = -1e30f;
        for (int c = lane; c < 1024; c += 32) {
            float p = Sr[c] * inv1 * add[c];
            Sr[c] = p;
            m2 = fmaxf(m2, p);
        }
        m2 = wred_max(m2);
        float s2 = 0.f;
        for (int c = lane; c < 1024; c += 32) { float e = __expf(Sr[c] - m2); Sr[c] = e; s2 += e; }
        s2 = wred_sum(s2);
        const float inv2 = __frcp_rn(s2);

        _Float16* Pr = P + r * PLD;
        for (int c = lane; c < 1024; c += 32) Pr[c] = (_Float16)(Sr[c] * inv2);
    }
    __syncthreads();

    // ---- out = P @ V  (wave w owns 16-wide d tile) ---------------------
    const int d0 = wave * 16;
    float8 acc = {};
    const _Float16* Vb = Vt + (size_t)((b * 16 + h) * 64 + d0) * 1024;
    for (int kc = 0; kc < 1024; kc += 32) {
        half16 a = frag_load(P + kc, PLD);   // A from LDS (ds_load_b128)
        half16 v = frag_load(Vb + kc, 1024); // B from transposed V
        acc = WMMA_F16(a, v, acc);
    }
#pragma unroll
    for (int r = 0; r < 8; ++r)
        AOut[(size_t)(b * 1024 + n0 + hi8 + r) * 1024 + h * 64 + d0 + sub] = (_Float16)acc[r];
}

// ---------------------------------------------------------------------------
// Host launcher
// ---------------------------------------------------------------------------
extern "C" void kernel_launch(void* const* d_in, const int* in_sizes, int n_in,
                              void* d_out, int out_size, void* d_ws, size_t ws_size,
                              hipStream_t stream) {
    (void)in_sizes; (void)n_in; (void)out_size; (void)ws_size;
    const float* x        = (const float*)d_in[0];  // [4,1024,1024]
    const float* k_in     = (const float*)d_in[1];  // [4,1024,1024]
    const float* attn_add = (const float*)d_in[2];  // [4,1024,1024]
    const float* Wq       = (const float*)d_in[3];  // [1024,1024]
    const float* Wkv      = (const float*)d_in[4];  // [1024,2048]
    const float* Wproj    = (const float*)d_in[5];  // [1024,1024]
    const float* bproj    = (const float*)d_in[6];  // [1024]
    float* out = (float*)d_out;

    char* ws = (char*)d_ws;
    const size_t MB = 1024ull * 1024ull;
    _Float16* Qh     = (_Float16*)(ws + 0 * MB);   // [4096][1024]
    _Float16* Kh     = (_Float16*)(ws + 8 * MB);   // [4096][1024]
    _Float16* Vt     = (_Float16*)(ws + 16 * MB);  // [B*H*64][1024]
    _Float16* AOut   = (_Float16*)(ws + 24 * MB);  // [4096][1024]
    _Float16* xh     = (_Float16*)(ws + 32 * MB);  // [4096][1024]
    _Float16* kh     = (_Float16*)(ws + 40 * MB);  // [4096][1024]
    _Float16* WqT    = (_Float16*)(ws + 48 * MB);  // [1024][1024]
    _Float16* WkvT   = (_Float16*)(ws + 50 * MB);  // [2048][1024]
    _Float16* WprojT = (_Float16*)(ws + 54 * MB);  // [1024][1024]

    const int NTOT = 4 * 1024 * 1024;  // elements in x / k_in

    // 1) one-time data prep (f16 convert + weight transpose)
    ca_cvt_f16<<<dim3((NTOT / 4 + 255) / 256), dim3(256), 0, stream>>>(x, xh, NTOT / 4);
    ca_cvt_f16<<<dim3((NTOT / 4 + 255) / 256), dim3(256), 0, stream>>>(k_in, kh, NTOT / 4);
    ca_transpose_f16<<<dim3(32, 32), dim3(32, 8), 0, stream>>>(Wq, WqT, 1024, 1024);
    ca_transpose_f16<<<dim3(64, 32), dim3(32, 8), 0, stream>>>(Wkv, WkvT, 1024, 2048);
    ca_transpose_f16<<<dim3(32, 32), dim3(32, 8), 0, stream>>>(Wproj, WprojT, 1024, 1024);

    // 2) projections: Q = xh @ Wq ; K = kh @ Wkv[:, :1024] ; V^T from Wkv[:, 1024:]
    ca_gemm_f16<<<dim3(32, 16), dim3(128), 0, stream>>>(xh, WqT, Qh, 1024, 1024, 1024, 1024);
    ca_gemm_f16<<<dim3(32, 16), dim3(128), 0, stream>>>(kh, WkvT, Kh, 1024, 1024, 1024, 1024);
    ca_gemm_vstore<<<dim3(32, 16), dim3(128), 0, stream>>>(kh, WkvT + 1024ull * 1024ull, Vt,
                                                           1024, 1024, 1024);

    // 3) fused attention (double softmax with attn_add modulation)
    const size_t smem = 16 * 1024 * 4 + 16 * PLD * 2;  // 98816 B
    ca_attention<<<dim3(4096), dim3(128), smem, stream>>>(Qh, Kh, Vt, attn_add, AOut);

    // 4) output projection + bias (f32 out)
    ca_gemm_f32_bias<<<dim3(32, 16), dim3(128), 0, stream>>>(AOut, WprojT, out, bproj,
                                                             1024, 1024, 1024, 1024);
}
